// WindowedMSA_90469191123273
// MI455X (gfx1250) — compile-verified, matrix operand order checked by reading, not run on hardware
//
#include <hip/hip_runtime.h>
#include <hip/hip_bf16.h>

typedef _Float16 half_t;
typedef __attribute__((ext_vector_type(16))) _Float16 v16h;
typedef __attribute__((ext_vector_type(8)))  float    v8f;

#define WIN    16
#define HEADS  4
#define CCH    96
#define HD     24
#define HDP    32
#define S_TOK  256
#define NWIN   512   // 2 * 16 * 16 windows
#define IMG_H  256
#define IMG_W  256

// -------- async global->LDS copy path (gfx1250 ASYNCcnt ops), guarded -------
#if defined(__has_builtin)
#  if __has_builtin(__builtin_amdgcn_global_load_async_to_lds_b128)
#    define USE_ASYNC_LDS 1
#  endif
#endif
#ifndef USE_ASYNC_LDS
#  define USE_ASYNC_LDS 0
#endif

#if USE_ASYNC_LDS
typedef int v4i_t __attribute__((ext_vector_type(4)));
typedef __attribute__((address_space(1))) v4i_t gv4i_t;   // global b128 chunk
typedef __attribute__((address_space(3))) v4i_t lv4i_t;   // LDS b128 chunk
template <int OFF>
__device__ inline void async_copy_b128(const void* gsrc, void* ldst) {
  __builtin_amdgcn_global_load_async_to_lds_b128((gv4i_t*)gsrc, (lv4i_t*)ldst,
                                                 OFF, 0);
}
__device__ inline void async_wait0() {
  asm volatile("s_wait_asynccnt 0x0" ::: "memory");
}
#endif

// ---- WMMA fragment loaders (wave32, 16-bit A/B 16x32 layout per CDNA5 ISA) ----
// Row-chunk pattern: lane r = lane&15 picks the row (A) / column-as-row (B=K^T),
// hi half of the wave takes K offsets +8 (VGPR0-3) and +24 (VGPR4-7).
__device__ inline v16h frag_row(const half_t* __restrict__ base, int row0, int stride, int k0) {
  const int lane = threadIdx.x & 31;
  const int r  = row0 + (lane & 15);
  const int hi = (lane >> 4) * 8;
  const half_t* p = base + r * stride + k0 + hi;
  v16h f;
#pragma unroll
  for (int j = 0; j < 8; ++j) { f[j] = p[j]; f[j + 8] = p[j + 16]; }
  return f;
}

// Column-gather pattern for B when the K dimension runs down rows of LDS
// (used for V in P@V): lane holds output column n, K strided by `stride`.
__device__ inline v16h frag_col(const half_t* __restrict__ base, int k0, int col0, int stride) {
  const int lane = threadIdx.x & 31;
  const int n  = col0 + (lane & 15);
  const int hi = (lane >> 4) * 8;
  v16h f;
#pragma unroll
  for (int j = 0; j < 8; ++j) {
    f[j]     = base[(k0 + hi + j) * stride + n];
    f[j + 8] = base[(k0 + hi + 16 + j) * stride + n];
  }
  return f;
}

// ================= Kernel 1: fused window-partition + QKV GEMM ================
// 1024 WGs x 256 thr. Each WG: 128 token rows; each of 8 waves one 16-row tile.
// LDS: z[128][96] f16 + Wt[288][96] f16 (transposed weights) = 78 KB (dynamic).
__global__ void qkv_kernel(const float* __restrict__ x,
                           const float* __restrict__ qkv_w,
                           const float* __restrict__ qkv_b,
                           half_t* __restrict__ qkv_ws) {
  extern __shared__ half_t sm1[];
  half_t* zt = sm1;                 // [128][96]
  half_t* wT = sm1 + 128 * CCH;     // [288][96]
  const int tid = threadIdx.x;
  const int rowBase = blockIdx.x * 128;

  // stage window-partitioned activations, fp32 -> f16 (needs convert: VALU path)
  for (int idx = tid; idx < 128 * CCH; idx += 256) {
    const int row = idx / CCH, col = idx - row * CCH;
    const int t = rowBase + row;
    const int n = t >> 8, s = t & 255;
    const int b = n >> 8, r = n & 255;
    const int wy = r >> 4, wx = r & 15;
    const int y = s >> 4, xx = s & 15;
    const size_t g = ((size_t)((b * IMG_H + wy * WIN + y) * IMG_W + wx * WIN + xx)) * CCH + col;
    zt[idx] = (half_t)x[g];
  }
  // stage transposed weights: wT[cc][kd] = qkv_w[kd][cc]
  for (int idx = tid; idx < CCH * 288; idx += 256) {
    const int kd = idx / 288, cc = idx - kd * 288;
    wT[cc * CCH + kd] = (half_t)qkv_w[idx];
  }
  __syncthreads();

  const int wave = tid >> 5, lane = tid & 31;
  const int m0 = wave * 16;
  const int colLane = lane & 15;
  const int rowOff = (lane >> 4) * 8;

  for (int ct = 0; ct < 18; ++ct) {            // 288 / 16 column tiles
    v8f acc = {};
#pragma unroll
    for (int kk = 0; kk < 3; ++kk) {           // K = 96
      v16h a = frag_row(zt, m0, CCH, kk * 32);
      v16h b = frag_row(wT, ct * 16, CCH, kk * 32);
      acc = __builtin_amdgcn_wmma_f32_16x16x32_f16(false, a, false, b,
                                                   (short)0, acc, false, false);
    }
    // channel cc -> (head, which q/k/v, d) per reference reshape (heads,3,hd)
    const int cc = ct * 16 + colLane;
    const int head = cc / 72;
    const int rem = cc - head * 72;
    const int which = rem / HD;
    const int dd = rem - which * HD;
    const float bias = qkv_b[cc];
#pragma unroll
    for (int i = 0; i < 8; ++i) {
      const int t = rowBase + m0 + i + rowOff;
      const int n = t >> 8, s = t & 255;
      const size_t o = ((((size_t)n * 3 + which) * HEADS + head) * S_TOK + s) * HD + dd;
      qkv_ws[o] = (half_t)(acc[i] + bias);
    }
  }
}

// ============ Kernel 2: per-(window, head) attention + LePE, fused ===========
// 2048 WGs x 256 thr. LDS: q/k/v [256][32] f16 (zero-padded hd 24->32) + 8
// per-wave [16][256] f16 prob tiles = 112 KB (dynamic). Each wave: 2 row tiles.
__global__ void attn_kernel(const half_t* __restrict__ qkv_ws,
                            const float* __restrict__ pe_w,
                            const float* __restrict__ pe_b,
                            half_t* __restrict__ o_ws) {
  extern __shared__ half_t sm2[];
  half_t* qs = sm2;                 // [256][32]
  half_t* ks = sm2 + 256 * HDP;
  half_t* vs = sm2 + 512 * HDP;
  half_t* pr = sm2 + 768 * HDP;     // 8 x [16][256]

  const int tid = threadIdx.x;
  const int n = blockIdx.x >> 2;
  const int head = blockIdx.x & 3;

  // stage q/k/v row s = tid (24 halves = 48B, 16B aligned) + zero pad to 32
  {
    const int s = tid;
    const uint4 z4 = make_uint4(0u, 0u, 0u, 0u);
#pragma unroll
    for (int which = 0; which < 3; ++which) {
      const size_t src = ((((size_t)n * 3 + which) * HEADS + head) * S_TOK + s) * HD;
      half_t* dbase = (which == 0) ? qs : ((which == 1) ? ks : vs);
      half_t* drow = dbase + s * HDP;
#if USE_ASYNC_LDS
      const half_t* srow = qkv_ws + src;
      async_copy_b128<0>(srow, drow);    // 48B payload as 3 x b128 async DMA
      async_copy_b128<16>(srow, drow);
      async_copy_b128<32>(srow, drow);
      *(uint4*)(drow + HD) = z4;         // zero pad d = 24..31
#else
      const uint4* sp = (const uint4*)(qkv_ws + src);
      uint4* dp = (uint4*)drow;
      dp[0] = sp[0]; dp[1] = sp[1]; dp[2] = sp[2]; dp[3] = z4;
#endif
    }
  }
#if USE_ASYNC_LDS
  async_wait0();
#endif
  __syncthreads();

  const int wave = tid >> 5, lane = tid & 31;
  half_t* prw = pr + wave * (16 * 256);
  const int colLane = lane & 15;
  const int rowOff = (lane >> 4) * 8;
  const float scale = 0.20412414523193154f;   // 1/sqrt(24)

  for (int pass = 0; pass < 2; ++pass) {
    const int m0 = (wave * 2 + pass) * 16;

    // S = Q K^T : full 16x256 row block in 16 accumulator tiles, single K-step
    v16h aq = frag_row(qs, m0, HDP, 0);
    v8f acc[16];
#pragma unroll
    for (int ct = 0; ct < 16; ++ct) {
      v16h bk = frag_row(ks, ct * 16, HDP, 0);   // B = K^T: lane = key token
      v8f z = {};
      acc[ct] = __builtin_amdgcn_wmma_f32_16x16x32_f16(false, aq, false, bk,
                                                       (short)0, z, false, false);
    }

    // row softmax: N lives in lanes (16-lane halves), tiles along registers
#pragma unroll
    for (int i = 0; i < 8; ++i) {
      float mx = -1e30f;
#pragma unroll
      for (int ct = 0; ct < 16; ++ct) { acc[ct][i] *= scale; mx = fmaxf(mx, acc[ct][i]); }
#pragma unroll
      for (int d = 8; d >= 1; d >>= 1) mx = fmaxf(mx, __shfl_xor(mx, d, 32));
      float sum = 0.f;
#pragma unroll
      for (int ct = 0; ct < 16; ++ct) {
        const float e = __expf(acc[ct][i] - mx);
        acc[ct][i] = e; sum += e;
      }
#pragma unroll
      for (int d = 8; d >= 1; d >>= 1) sum += __shfl_xor(sum, d, 32);
      const float inv = 1.0f / sum;
#pragma unroll
      for (int ct = 0; ct < 16; ++ct) acc[ct][i] *= inv;
    }

    // re-fragment probs through private LDS tile (C-layout -> A-layout)
#pragma unroll
    for (int ct = 0; ct < 16; ++ct)
#pragma unroll
      for (int i = 0; i < 8; ++i)
        prw[(i + rowOff) * 256 + ct * 16 + colLane] = (half_t)acc[ct][i];

    // O = P V : [16x256] @ [256x32], K = 256 in 8 steps
    v8f oacc[2] = {};
#pragma unroll
    for (int kk = 0; kk < 8; ++kk) {
      v16h ap = frag_row(prw, 0, 256, kk * 32);
#pragma unroll
      for (int nt = 0; nt < 2; ++nt) {
        v16h bv = frag_col(vs, kk * 32, nt * 16, HDP);
        oacc[nt] = __builtin_amdgcn_wmma_f32_16x16x32_f16(false, ap, false, bv,
                                                          (short)0, oacc[nt], false, false);
      }
    }

    // fused LePE (3x3 depthwise, zero pad at window edge) + store f16 o
#pragma unroll
    for (int nt = 0; nt < 2; ++nt) {
      const int d = nt * 16 + colLane;
      if (d < HD) {
        const int c = head * HD + d;
        const float pb = pe_b[c];
        float w9[9];
#pragma unroll
        for (int kxy = 0; kxy < 9; ++kxy) w9[kxy] = pe_w[kxy * CCH + c];
#pragma unroll
        for (int i = 0; i < 8; ++i) {
          const int srow = m0 + i + rowOff;
          const int y = srow >> 4, xx = srow & 15;
          float l = pb;
#pragma unroll
          for (int dy = -1; dy <= 1; ++dy)
#pragma unroll
            for (int dx = -1; dx <= 1; ++dx) {
              const int yy = y + dy, xc = xx + dx;
              if (yy >= 0 && yy < WIN && xc >= 0 && xc < WIN)
                l += (float)vs[(yy * WIN + xc) * HDP + d] * w9[(dy + 1) * 3 + (dx + 1)];
            }
          o_ws[((size_t)n * S_TOK + srow) * CCH + c] = (half_t)(oacc[nt][i] + l);
        }
      }
    }
  }
}

// ======= Kernel 3: out projection + reverse window partition (fp32 out) ======
__global__ void proj_kernel(const half_t* __restrict__ o_ws,
                            const float* __restrict__ out_w,
                            const float* __restrict__ out_b,
                            float* __restrict__ out) {
  __shared__ __align__(16) half_t ot[128 * CCH];   // 24 KB
  __shared__ __align__(16) half_t wT[CCH * CCH];   // 18 KB
  const int tid = threadIdx.x;
  const int rowBase = blockIdx.x * 128;

#if USE_ASYNC_LDS
  // pure f16 memcpy of the 24KB activation block: async DMA, 16B chunks
  {
    const half_t* src = o_ws + (size_t)rowBase * CCH;
    for (int ch = tid; ch < (128 * CCH) / 8; ch += 256)
      async_copy_b128<0>(src + ch * 8, ot + ch * 8);
  }
#else
  for (int idx = tid; idx < 128 * CCH; idx += 256)
    ot[idx] = o_ws[(size_t)rowBase * CCH + idx];
#endif
  for (int idx = tid; idx < CCH * CCH; idx += 256) {
    const int kd = idx / CCH, cc = idx - kd * CCH;
    wT[cc * CCH + kd] = (half_t)out_w[idx];
  }
#if USE_ASYNC_LDS
  async_wait0();
#endif
  __syncthreads();

  const int wave = tid >> 5, lane = tid & 31;
  const int m0 = wave * 16;
  const int colLane = lane & 15;
  const int rowOff = (lane >> 4) * 8;

  for (int ct = 0; ct < 6; ++ct) {             // 96 / 16 column tiles
    v8f acc = {};
#pragma unroll
    for (int kk = 0; kk < 3; ++kk) {           // K = 96
      v16h a = frag_row(ot, m0, CCH, kk * 32);
      v16h b = frag_row(wT, ct * 16, CCH, kk * 32);
      acc = __builtin_amdgcn_wmma_f32_16x16x32_f16(false, a, false, b,
                                                   (short)0, acc, false, false);
    }
    const int c = ct * 16 + colLane;
    const float bias = out_b[c];
#pragma unroll
    for (int i = 0; i < 8; ++i) {
      const int t = rowBase + m0 + i + rowOff;
      const int n = t >> 8, s = t & 255;
      const int b = n >> 8, r = n & 255;
      const int wy = r >> 4, wx = r & 15;
      const int y = s >> 4, xx = s & 15;
      const size_t g = ((size_t)((b * IMG_H + wy * WIN + y) * IMG_W + wx * WIN + xx)) * CCH + c;
      out[g] = acc[i] + bias;
    }
  }
}

extern "C" void kernel_launch(void* const* d_in, const int* in_sizes, int n_in,
                              void* d_out, int out_size, void* d_ws, size_t ws_size,
                              hipStream_t stream) {
  (void)in_sizes; (void)n_in; (void)out_size; (void)ws_size;
  const float* x     = (const float*)d_in[0];
  const float* qkv_w = (const float*)d_in[1];
  const float* qkv_b = (const float*)d_in[2];
  const float* pe_w  = (const float*)d_in[3];
  const float* pe_b  = (const float*)d_in[4];
  const float* out_w = (const float*)d_in[5];
  const float* out_b = (const float*)d_in[6];
  float* out = (float*)d_out;

  // workspace: qkv f16 [512][3][4][256][24] (75.5 MB) + o f16 [512][256][96] (25 MB)
  half_t* qkv_ws = (half_t*)d_ws;
  half_t* o_ws   = qkv_ws + (size_t)NWIN * 3 * HEADS * S_TOK * HD;

  const int lds1 = (128 * CCH + 288 * CCH) * (int)sizeof(half_t);   //  79.9 KB
  const int lds2 = (768 * HDP + 8 * 16 * 256) * (int)sizeof(half_t); // 112 KB
  (void)hipFuncSetAttribute(reinterpret_cast<const void*>(qkv_kernel),
                            hipFuncAttributeMaxDynamicSharedMemorySize, lds1);
  (void)hipFuncSetAttribute(reinterpret_cast<const void*>(attn_kernel),
                            hipFuncAttributeMaxDynamicSharedMemorySize, lds2);

  qkv_kernel<<<1024, 256, lds1, stream>>>(x, qkv_w, qkv_b, qkv_ws);
  attn_kernel<<<NWIN * HEADS, 256, lds2, stream>>>(qkv_ws, pe_w, pe_b, o_ws);
  proj_kernel<<<1024, 256, 0, stream>>>(o_ws, out_w, out_b, out);
}
